// WaveNet_76871324664522
// MI455X (gfx1250) — matrix operand order repaired and checked by reading, not compiled, not run on hardware
//
#include <hip/hip_runtime.h>
#include <hip/hip_bf16.h>
#include <math.h>

typedef __attribute__((ext_vector_type(16))) _Float16 v16h;
typedef __attribute__((ext_vector_type(8)))  _Float16 v8h;
typedef __attribute__((ext_vector_type(8)))  float    v8f;

// ---- problem constants (match reference setup_inputs) ----
#define T_FULL 2048
#define BATCH  16
#define DMODEL 256
#define NLAY   8
#define LOUT   96
#define WIN    352      // trailing window of timesteps we materialize (>= 351 receptive field)
#define HP     520      // LDS row pitch (halfs) for K=512 A staging (bank-conflict pad)
#define GP     264      // LDS row pitch (halfs) for K=256 A staging
#define TMB    2        // timesteps (M-tiles) per workgroup: B-fragment reuse x2

// ---- workspace layout (bytes) ----
#define OFF_H     0
#define SZ_H      (WIN*BATCH*DMODEL*4)
#define OFF_G     (OFF_H + SZ_H)
#define SZ_G      (WIN*BATCH*DMODEL*4)
#define OFF_SKIP  (OFF_G + SZ_G)
#define SZ_SKIP   (LOUT*BATCH*DMODEL*4)
#define OFF_BC    (OFF_SKIP + SZ_SKIP)         // packed f16 dilated-conv weights
#define SZ_BC     (NLAY*512*512*2)
#define OFF_BS    (OFF_BC + SZ_BC)             // packed f16 skip-conv weights
#define SZ_BS     (NLAY*256*512*2)

// ---------------------------------------------------------------------------
// Weight repack: f32 -> f16 in WMMA-B-fragment-native tiles.
// Tile = 32(K) x 16(N); lane l owns col n=l&15, K-rows (l<16 ? 0..15 : 16..31);
// each lane's 16 halfs contiguous => B fragment = one contiguous 32B load/lane.
// ---------------------------------------------------------------------------
__global__ void pack_conv_w(const float* __restrict__ dil_w, _Float16* __restrict__ bc) {
  int idx = blockIdx.x * 256 + threadIdx.x;          // NLAY*512*512
  if (idx >= NLAY * 512 * 512) return;
  int i  = idx >> 18;
  int r  = idx & (512 * 512 - 1);
  int kk = r >> 9;                                   // 0..511 : [h(t-dil) ch | h(t) ch]
  int o  = r & 511;                                  // output channel
  int k  = kk >> 8;                                  // conv tap (0 -> t-dil, 1 -> t)
  int c  = kk & 255;                                 // input channel
  float v = dil_w[(((i * 512 + o) * 256 + c) << 1) + k];
  int kt = kk >> 5, ot = o >> 4;
  int l  = (o & 15) + (((kk >> 4) & 1) << 4);
  int j  = kk & 15;
  bc[(((i * 16 + kt) * 32 + ot) << 9) + (l << 4) + j] = (_Float16)v;
}

__global__ void pack_skip_w(const float* __restrict__ skip_w, _Float16* __restrict__ bs) {
  int idx = blockIdx.x * 256 + threadIdx.x;          // NLAY*256*512
  if (idx >= NLAY * 256 * 512) return;
  int i  = idx >> 17;
  int r  = idx & (256 * 512 - 1);
  int kk = r >> 9;                                   // input channel 0..255
  int o  = r & 511;                                  // output channel
  float v = skip_w[(i * 512 + o) * 256 + kk];
  int kt = kk >> 5, ot = o >> 4;
  int l  = (o & 15) + (((kk >> 4) & 1) << 4);
  int j  = kk & 15;
  bs[(((i * 8 + kt) * 32 + ot) << 9) + (l << 4) + j] = (_Float16)v;
}

// ---------------------------------------------------------------------------
// Upsample: h[tl,b,c] = [x_lag | x_cov] @ up_W + up_b  (8 -> 256), trailing WIN steps
// ---------------------------------------------------------------------------
__global__ void upsample_kernel(const float* __restrict__ xlag, const float* __restrict__ xcov,
                                const float* __restrict__ upW, const float* __restrict__ upb,
                                float* __restrict__ h) {
  __shared__ float vin[8];
  int blk = blockIdx.x;                 // tl*16 + b
  int tl = blk >> 4, b = blk & 15;
  int t = T_FULL - WIN + tl;
  int tid = threadIdx.x;
  if (tid == 0)      vin[0]   = xlag[t * BATCH + b];
  else if (tid < 8)  vin[tid] = xcov[(t * BATCH + b) * 7 + (tid - 1)];
  __syncthreads();
  float acc = upb[tid];
#pragma unroll
  for (int j = 0; j < 8; ++j) acc += vin[j] * upW[j * DMODEL + tid];
  h[(tl * BATCH + b) * DMODEL + tid] = acc;
}

__global__ void zero_kernel(float* __restrict__ p, int n) {
  int i = blockIdx.x * 256 + threadIdx.x;
  if (i < n) p[i] = 0.0f;
}

// ---------------------------------------------------------------------------
// Gated dilated conv: one block per 2 timesteps; GEMM (2x M=16) K=512 N=512,
// fused bias + tanh(f)*sigmoid(g). 8 waves x (2 ts x 4 subtiles) accumulators,
// 16 K-steps, each B fragment reused by both timesteps' A fragments.
// ---------------------------------------------------------------------------
__global__ __launch_bounds__(256)
void gated_kernel(const float* __restrict__ h, const v16h* __restrict__ Bc,
                  const float* __restrict__ bias, float* __restrict__ gated,
                  int dil, int t0) {
  __shared__ _Float16 lds[TMB * 16 * HP];
  int tl0 = t0 + TMB * blockIdx.x;
  int tid = threadIdx.x;
  // Stage A rows for both timesteps: kk<256 -> h[t-dil], kk>=256 -> h[t]
  for (int idx = tid; idx < TMB * 16 * 512; idx += 256) {
    int ts = idx >> 13;
    int r  = idx & 8191;
    int b  = r >> 9, kk = r & 511;
    int tl = tl0 + ts, tm = tl - dil;
    float v = (kk < 256) ? h[(tm * BATCH + b) * DMODEL + kk]
                         : h[(tl * BATCH + b) * DMODEL + (kk - 256)];
    lds[(ts * 16 + b) * HP + kk] = (_Float16)v;
  }
  __syncthreads();

  int wave = tid >> 5, lane = tid & 31;
  int ob = wave * 32;                      // this wave's 32 gated channels
  int n = lane & 15;
  int kbase = (lane >> 4) << 4;            // A-fragment K sub-offset (contiguous 16 per lane half)
  const _Float16* ap0 = &lds[n * HP];
  const _Float16* ap1 = &lds[(16 + n) * HP];
  int ot0 = ob >> 4;                       // f subtile col-tile
  int ot2 = (256 + ob) >> 4;               // g subtile col-tile

  v8f c[TMB][4] = {};
  for (int kt = 0; kt < 16; ++kt) {
    int k0 = kt << 5;
    v8h lo0 = *(const v8h*)(ap0 + k0 + kbase);
    v8h hi0 = *(const v8h*)(ap0 + k0 + kbase + 8);
    v8h lo1 = *(const v8h*)(ap1 + k0 + kbase);
    v8h hi1 = *(const v8h*)(ap1 + k0 + kbase + 8);
    v16h a0 = __builtin_shufflevector(lo0, hi0, 0,1,2,3,4,5,6,7,8,9,10,11,12,13,14,15);
    v16h a1 = __builtin_shufflevector(lo1, hi1, 0,1,2,3,4,5,6,7,8,9,10,11,12,13,14,15);
    v16h b0 = Bc[((kt * 32 + ot0    ) << 5) + lane];
    v16h b1 = Bc[((kt * 32 + ot0 + 1) << 5) + lane];
    v16h b2 = Bc[((kt * 32 + ot2    ) << 5) + lane];
    v16h b3 = Bc[((kt * 32 + ot2 + 1) << 5) + lane];
    c[0][0] = __builtin_amdgcn_wmma_f32_16x16x32_f16(false, a0, false, b0, (short)0, c[0][0], false, false);
    c[1][0] = __builtin_amdgcn_wmma_f32_16x16x32_f16(false, a1, false, b0, (short)0, c[1][0], false, false);
    c[0][1] = __builtin_amdgcn_wmma_f32_16x16x32_f16(false, a0, false, b1, (short)0, c[0][1], false, false);
    c[1][1] = __builtin_amdgcn_wmma_f32_16x16x32_f16(false, a1, false, b1, (short)0, c[1][1], false, false);
    c[0][2] = __builtin_amdgcn_wmma_f32_16x16x32_f16(false, a0, false, b2, (short)0, c[0][2], false, false);
    c[1][2] = __builtin_amdgcn_wmma_f32_16x16x32_f16(false, a1, false, b2, (short)0, c[1][2], false, false);
    c[0][3] = __builtin_amdgcn_wmma_f32_16x16x32_f16(false, a0, false, b3, (short)0, c[0][3], false, false);
    c[1][3] = __builtin_amdgcn_wmma_f32_16x16x32_f16(false, a1, false, b3, (short)0, c[1][3], false, false);
  }

  int mbase = (lane >> 4) << 3;            // C-fragment M sub-offset
  float bf0 = bias[ob + n],        bf1 = bias[ob + 16 + n];
  float bg0 = bias[256 + ob + n],  bg1 = bias[256 + ob + 16 + n];
#pragma unroll
  for (int ts = 0; ts < TMB; ++ts) {
    int tl = tl0 + ts;
#pragma unroll
    for (int v = 0; v < 8; ++v) {
      int b = v + mbase;
      float f0 = c[ts][0][v] + bf0, g0 = c[ts][2][v] + bg0;
      float f1 = c[ts][1][v] + bf1, g1 = c[ts][3][v] + bg1;
      float r0 = tanhf(f0) * (1.0f / (1.0f + __expf(-g0)));
      float r1 = tanhf(f1) * (1.0f / (1.0f + __expf(-g1)));
      gated[(tl * BATCH + b) * DMODEL + ob + n]      = r0;
      gated[(tl * BATCH + b) * DMODEL + ob + 16 + n] = r1;
    }
  }
}

// ---------------------------------------------------------------------------
// 1x1 skip conv: one block per 2 timesteps; GEMM (2x M=16) K=256 N=512;
// waves 0-3 accumulate into h (residual), waves 4-7 into skip (last LOUT steps).
// ---------------------------------------------------------------------------
__global__ __launch_bounds__(256)
void skipconv_kernel(const float* __restrict__ gated, const v16h* __restrict__ Bs,
                     const float* __restrict__ bias, float* __restrict__ h,
                     float* __restrict__ skip, int t0) {
  __shared__ _Float16 lds[TMB * 16 * GP];
  int tl0 = t0 + TMB * blockIdx.x;
  int tid = threadIdx.x;
  for (int idx = tid; idx < TMB * 16 * 256; idx += 256) {
    int ts = idx >> 12;
    int r  = idx & 4095;
    int b  = r >> 8, kk = r & 255;
    lds[(ts * 16 + b) * GP + kk] = (_Float16)gated[((tl0 + ts) * BATCH + b) * DMODEL + kk];
  }
  __syncthreads();

  int wave = tid >> 5, lane = tid & 31;
  int ob = wave * 64;                      // 64 output cols of 512
  int n = lane & 15;
  int kbase = (lane >> 4) << 4;
  const _Float16* ap0 = &lds[n * GP];
  const _Float16* ap1 = &lds[(16 + n) * GP];

  v8f acc[TMB][4] = {};
  for (int kt = 0; kt < 8; ++kt) {
    int k0 = kt << 5;
    v8h lo0 = *(const v8h*)(ap0 + k0 + kbase);
    v8h hi0 = *(const v8h*)(ap0 + k0 + kbase + 8);
    v8h lo1 = *(const v8h*)(ap1 + k0 + kbase);
    v8h hi1 = *(const v8h*)(ap1 + k0 + kbase + 8);
    v16h a0 = __builtin_shufflevector(lo0, hi0, 0,1,2,3,4,5,6,7,8,9,10,11,12,13,14,15);
    v16h a1 = __builtin_shufflevector(lo1, hi1, 0,1,2,3,4,5,6,7,8,9,10,11,12,13,14,15);
#pragma unroll
    for (int s = 0; s < 4; ++s) {
      v16h bm = Bs[((kt * 32 + (ob >> 4) + s) << 5) + lane];
      acc[0][s] = __builtin_amdgcn_wmma_f32_16x16x32_f16(false, a0, false, bm, (short)0, acc[0][s], false, false);
      acc[1][s] = __builtin_amdgcn_wmma_f32_16x16x32_f16(false, a1, false, bm, (short)0, acc[1][s], false, false);
    }
  }

  int mbase = (lane >> 4) << 3;
#pragma unroll
  for (int ts = 0; ts < TMB; ++ts) {
    int tl = tl0 + ts;
#pragma unroll
    for (int s = 0; s < 4; ++s) {
      int o = ob + s * 16 + n;
      float bb = bias[o];
#pragma unroll
      for (int v = 0; v < 8; ++v) {
        int b = v + mbase;
        float val = acc[ts][s][v] + bb;
        if (o < 256) {                                 // residual half (uniform per wave)
          int idx = (tl * BATCH + b) * DMODEL + o;
          h[idx] += val;
        } else if (tl >= WIN - LOUT) {                 // skip half, last LOUT steps
          int tss = tl - (WIN - LOUT);
          skip[(tss * BATCH + b) * DMODEL + (o - 256)] += val;
        }
      }
    }
  }
}

// ---------------------------------------------------------------------------
// Head: y[b,l] = (skip[l,b,:] . loc_W + loc_b) * proj_W + proj_b
// ---------------------------------------------------------------------------
__global__ void head_kernel(const float* __restrict__ skip, const float* __restrict__ locW,
                            const float* __restrict__ locb, const float* __restrict__ projW,
                            const float* __restrict__ projb, float* __restrict__ out) {
  int tid = blockIdx.x * 256 + threadIdx.x;
  if (tid >= BATCH * LOUT) return;
  int b = tid / LOUT, l = tid % LOUT;
  float acc = 0.0f;
  for (int c = 0; c < DMODEL; ++c) acc += skip[(l * BATCH + b) * DMODEL + c] * locW[c];
  out[b * LOUT + l] = (acc + locb[0]) * projW[0] + projb[0];
}

// ---------------------------------------------------------------------------
extern "C" void kernel_launch(void* const* d_in, const int* in_sizes, int n_in,
                              void* d_out, int out_size, void* d_ws, size_t ws_size,
                              hipStream_t stream) {
  (void)in_sizes; (void)n_in; (void)out_size; (void)ws_size;
  const float* x_lag  = (const float*)d_in[0];
  const float* x_cov  = (const float*)d_in[1];
  const float* up_W   = (const float*)d_in[2];
  const float* up_b   = (const float*)d_in[3];
  const float* dil_w  = (const float*)d_in[4];
  const float* dil_b  = (const float*)d_in[5];
  const float* skip_w = (const float*)d_in[6];
  const float* skip_b = (const float*)d_in[7];
  const float* loc_W  = (const float*)d_in[8];
  const float* loc_b  = (const float*)d_in[9];
  const float* proj_W = (const float*)d_in[10];
  const float* proj_b = (const float*)d_in[11];

  char* ws = (char*)d_ws;
  float*     h     = (float*)(ws + OFF_H);
  float*     gated = (float*)(ws + OFF_G);
  float*     skip  = (float*)(ws + OFF_SKIP);
  _Float16*  Bc    = (_Float16*)(ws + OFF_BC);
  _Float16*  Bs    = (_Float16*)(ws + OFF_BS);

  pack_conv_w<<<(NLAY * 512 * 512 + 255) / 256, 256, 0, stream>>>(dil_w, Bc);
  pack_skip_w<<<(NLAY * 256 * 512 + 255) / 256, 256, 0, stream>>>(skip_w, Bs);
  zero_kernel<<<(LOUT * BATCH * DMODEL + 255) / 256, 256, 0, stream>>>(skip, LOUT * BATCH * DMODEL);
  upsample_kernel<<<WIN * BATCH, 256, 0, stream>>>(x_lag, x_cov, up_W, up_b, h);

  // Exact shrinking receptive-field windows (all even -> TMB=2 tiles divide exactly).
  const int Wnext[NLAY] = {350, 348, 344, 336, 320, 288, 224, 96};
  for (int i = 0; i < NLAY; ++i) {
    int nt  = Wnext[i];
    int t0  = WIN - nt;
    int dil = 1 << i;
    gated_kernel<<<nt / TMB, 256, 0, stream>>>(h, (const v16h*)(Bc + (size_t)i * 16 * 32 * 512),
                                               dil_b + i * 512, gated, dil, t0);
    skipconv_kernel<<<nt / TMB, 256, 0, stream>>>(gated, (const v16h*)(Bs + (size_t)i * 8 * 32 * 512),
                                                  skip_b + i * 512, h, skip, t0);
  }

  head_kernel<<<(BATCH * LOUT + 255) / 256, 256, 0, stream>>>(skip, loc_W, loc_b, proj_W, proj_b,
                                                              (float*)d_out);
}
